// STSGCM_66632122630764
// MI455X (gfx1250) — compile-verified, matrix operand order checked by reading, not run on hardware
//
#include <hip/hip_runtime.h>

#define N3   3072   // 3 * N_SERIES
#define NB   64     // batch
#define NC   64     // channels / h_dim
#define NMID0 1024
#define NMID1 2048

typedef __attribute__((ext_vector_type(16))) __bf16 v16bf;
typedef __attribute__((ext_vector_type(8)))  float  v8f;
typedef __attribute__((ext_vector_type(4)))  int    v4i;

struct alignas(16) U128 { unsigned int x[4]; };
union FragU { U128 q[2]; v16bf v; };

#if __has_builtin(__builtin_amdgcn_global_load_async_to_lds_b128) && \
    __has_builtin(__builtin_amdgcn_s_wait_asynccnt)
#define USE_ASYNC_LDS 1
#else
#define USE_ASYNC_LDS 0
#endif

#if USE_ASYNC_LDS
#define AS1 __attribute__((address_space(1)))
#define AS3 __attribute__((address_space(3)))
static __device__ __forceinline__ void async_ld16(const void* g, void* l) {
  // one b128 per lane, memory -> LDS, tracked by ASYNCcnt
  __builtin_amdgcn_global_load_async_to_lds_b128(
      (AS1 v4i*)g, (AS3 v4i*)l, 0, 0);
}
#define WAIT_ASYNC(n) __builtin_amdgcn_s_wait_asynccnt(n)
#else
#define WAIT_ASYNC(n)
#endif

static __device__ __forceinline__ unsigned short f2bf(float f) {
  unsigned int u = __builtin_bit_cast(unsigned int, f);
  u += 0x7FFFu + ((u >> 16) & 1u);        // round-to-nearest-even
  return (unsigned short)(u >> 16);
}

// ---------------- conversion kernels ----------------

__global__ __launch_bounds__(256) void cvt_bf16_kernel(const float* __restrict__ in,
                                                       unsigned short* __restrict__ out,
                                                       int n) {
  int i = blockIdx.x * 256 + threadIdx.x;
  if (i < n) out[i] = f2bf(in[i]);
}

// x: (3N, B*C) f32  ->  H: (B*C, 3N) bf16   (LDS-tiled transpose)
__global__ __launch_bounds__(256) void cvt_x_kernel(const float* __restrict__ x,
                                                    unsigned short* __restrict__ H) {
  __shared__ float tile[32][33];
  const int tx = threadIdx.x & 31;
  const int ty = threadIdx.x >> 5;          // 0..7
  const int m0 = blockIdx.x * 32;           // column (b*64+c) tile
  const int w0 = blockIdx.y * 32;           // row (node) tile
#pragma unroll
  for (int i = 0; i < 32; i += 8)
    tile[ty + i][tx] = x[(size_t)(w0 + ty + i) * (NB * NC) + m0 + tx];
  __syncthreads();
#pragma unroll
  for (int i = 0; i < 32; i += 8)
    H[(size_t)(m0 + ty + i) * N3 + w0 + tx] = f2bf(tile[tx][ty + i]);
}

// ---------------- fused layer kernel ----------------
// block = 128 threads (4 waves). Output tile: M = 128 rows (2 batches) x V = 64.
// Each wave: 2 M-strips (one per batch) x 4 v-subtiles = 8 WMMA accumulators.
// Double-buffered LDS, async global->LDS staging (ASYNCcnt).
__global__ __launch_bounds__(128) void stsgcm_layer_kernel(
    const unsigned short* __restrict__ Hin,   // (4096, 3072) bf16
    const unsigned short* __restrict__ Abf,   // (3072, 3072) bf16, A[v][w]
    const unsigned short* __restrict__ Wbf,   // (128, 64) bf16
    const float* __restrict__ bias,           // (128,) f32
    unsigned short* __restrict__ Hout,        // (4096, 3072) bf16 (depth 0 only)
    float* __restrict__ dout,                 // (1024, 64, 64) f32
    int depth) {
  __shared__ alignas(16) unsigned short Ht[2][128][72];  // H tiles (ping/pong); [0] reused as Gt
  __shared__ alignas(16) unsigned short At[2][64][72];   // A tiles (ping/pong)

  const int tid   = threadIdx.x;
  const int lane  = tid & 31;
  const int wave  = tid >> 5;                 // 0..3
  const int lr    = lane & 15;                // row/col within 16
  const int lh    = (lane >> 4) * 8;          // K-half select: 0 or 8
  const int v0    = blockIdx.x * 64;
  const int mb128 = blockIdx.y * 128;         // first of 2 batches * 64

  // staging geometry: thread covers rows row0 + i*16, fixed 16B column chunk
  const int row0 = tid >> 3;                  // 0..15
  const int col0 = (tid & 7) * 8;             // element offset (16B chunks)

  v8f acc0[4] = {}, acc1[4] = {};

  constexpr int NCHUNK = N3 / 64;             // 48

  auto stage = [&](int c, int buf) {
    const unsigned short* gh = Hin + (size_t)(mb128 + row0) * N3 + c * 64 + col0;
    const unsigned short* ga = Abf + (size_t)(v0 + row0) * N3 + c * 64 + col0;
#if USE_ASYNC_LDS
#pragma unroll
    for (int i = 0; i < 8; ++i)
      async_ld16(gh + (size_t)i * 16 * N3, &Ht[buf][row0 + i * 16][col0]);
#pragma unroll
    for (int i = 0; i < 4; ++i)
      async_ld16(ga + (size_t)i * 16 * N3, &At[buf][row0 + i * 16][col0]);
#else
#pragma unroll
    for (int i = 0; i < 8; ++i)
      *(U128*)&Ht[buf][row0 + i * 16][col0] = *(const U128*)(gh + (size_t)i * 16 * N3);
#pragma unroll
    for (int i = 0; i < 4; ++i)
      *(U128*)&At[buf][row0 + i * 16][col0] = *(const U128*)(ga + (size_t)i * 16 * N3);
#endif
  };

  stage(0, 0);
  int cur = 0;
  for (int c = 0; c < NCHUNK; ++c) {
    if (c + 1 < NCHUNK) {
      stage(c + 1, cur ^ 1);                  // prefetch next chunk into other buffer
      WAIT_ASYNC(12);                         // retire chunk c (12 ops), keep c+1 in flight
    } else {
      WAIT_ASYNC(0);
    }
    __syncthreads();                          // chunk c visible to all waves

    const unsigned short (*bH)[72] = Ht[cur];
    const unsigned short (*bA)[72] = At[cur];
#pragma unroll
    for (int kk = 0; kk < 64; kk += 32) {
      FragU fa0, fa1;                         // A-frags: batch0 / batch1 M-strips
      fa0.q[0] = *(const U128*)&bH[wave * 16 + lr][kk + lh];
      fa0.q[1] = *(const U128*)&bH[wave * 16 + lr][kk + lh + 16];
      fa1.q[0] = *(const U128*)&bH[64 + wave * 16 + lr][kk + lh];
      fa1.q[1] = *(const U128*)&bH[64 + wave * 16 + lr][kk + lh + 16];
#pragma unroll
      for (int vs = 0; vs < 4; ++vs) {
        FragU fb;                             // B-frag shared by both M-strips
        fb.q[0] = *(const U128*)&bA[vs * 16 + lr][kk + lh];
        fb.q[1] = *(const U128*)&bA[vs * 16 + lr][kk + lh + 16];
        acc0[vs] = __builtin_amdgcn_wmma_f32_16x16x32_bf16(
            false, fa0.v, false, fb.v, (short)0, acc0[vs], false, false);
        acc1[vs] = __builtin_amdgcn_wmma_f32_16x16x32_bf16(
            false, fa1.v, false, fb.v, (short)0, acc1[vs], false, false);
      }
    }
    __syncthreads();                          // done reading cur before it is refilled
    cur ^= 1;
  }

  // ---- spill G transposed to LDS as bf16: Gt[b][v][c] packed in Ht[0] ----
#pragma unroll
  for (int vs = 0; vs < 4; ++vs)
#pragma unroll
    for (int r = 0; r < 8; ++r) {
      int cch = wave * 16 + r + lh;           // channel row of this accum element
      Ht[0][vs * 16 + lr][cch]      = f2bf(acc0[vs][r]);
      Ht[0][64 + vs * 16 + lr][cch] = f2bf(acc1[vs][r]);
    }
  __syncthreads();

  // ---- epilogue per batch: Y = W(128x64) @ G(64x64), bias + GLU, stores ----
#pragma unroll
  for (int bb = 0; bb < 2; ++bb) {
    v8f yl[4] = {}, yr[4] = {};
#pragma unroll
    for (int kk = 0; kk < 64; kk += 32) {
      FragU wl, wr;
      const int orow = wave * 16 + lr;        // lhs rows o = 0..63, rhs o = 64..127
      wl.q[0] = *(const U128*)(Wbf + orow * 64 + kk + lh);
      wl.q[1] = *(const U128*)(Wbf + orow * 64 + kk + lh + 16);
      wr.q[0] = *(const U128*)(Wbf + (orow + 64) * 64 + kk + lh);
      wr.q[1] = *(const U128*)(Wbf + (orow + 64) * 64 + kk + lh + 16);
#pragma unroll
      for (int vs = 0; vs < 4; ++vs) {
        FragU g;                              // B-frag col v: Gt[v][k] contiguous
        g.q[0] = *(const U128*)&Ht[0][bb * 64 + vs * 16 + lr][kk + lh];
        g.q[1] = *(const U128*)&Ht[0][bb * 64 + vs * 16 + lr][kk + lh + 16];
        yl[vs] = __builtin_amdgcn_wmma_f32_16x16x32_bf16(
            false, wl.v, false, g.v, (short)0, yl[vs], false, false);
        yr[vs] = __builtin_amdgcn_wmma_f32_16x16x32_bf16(
            false, wr.v, false, g.v, (short)0, yr[vs], false, false);
      }
    }
    const int bidx = blockIdx.y * 2 + bb;
#pragma unroll
    for (int vs = 0; vs < 4; ++vs) {
      const int v = v0 + vs * 16 + lr;
#pragma unroll
      for (int r = 0; r < 8; ++r) {
        const int cch = wave * 16 + r + lh;   // 0..63
        float a = yl[vs][r] + bias[cch];
        float g = yr[vs][r] + bias[cch + 64];
        float h = a * (1.0f / (1.0f + __expf(-g)));
        if (depth == 0)
          Hout[(size_t)(bidx * 64 + cch) * N3 + v] = f2bf(h);
        if (v >= NMID0 && v < NMID1) {
          float* op = dout + ((size_t)(v - NMID0) * NB + bidx) * NC + cch;
          *op = (depth == 0) ? h : fmaxf(*op, h);
        }
      }
    }
  }
}

// ---------------- host launcher ----------------

extern "C" void kernel_launch(void* const* d_in, const int* in_sizes, int n_in,
                              void* d_out, int out_size, void* d_ws, size_t ws_size,
                              hipStream_t stream) {
  (void)in_sizes; (void)n_in; (void)out_size; (void)ws_size;
  const float* x  = (const float*)d_in[0];   // (3072, 64, 64)
  const float* A  = (const float*)d_in[1];   // (3072, 3072)
  const float* W0 = (const float*)d_in[2];   // (128, 64)
  const float* b0 = (const float*)d_in[3];   // (128,)
  const float* W1 = (const float*)d_in[4];
  const float* b1 = (const float*)d_in[5];
  float* out = (float*)d_out;                // (1024, 64, 64)

  char* ws = (char*)d_ws;
  const size_t A_BYTES = (size_t)N3 * N3 * 2;        // 18,874,368
  const size_t H_BYTES = (size_t)NB * NC * N3 * 2;   // 25,165,824
  unsigned short* Abf = (unsigned short*)ws;
  unsigned short* H0  = (unsigned short*)(ws + A_BYTES);
  unsigned short* H1  = (unsigned short*)(ws + A_BYTES + H_BYTES);
  unsigned short* Wb  = (unsigned short*)(ws + A_BYTES + 2 * H_BYTES);

  // convert A, W0, W1 to bf16; transpose+convert x to (b*64+c, w) bf16
  const int nA = N3 * N3;
  cvt_bf16_kernel<<<(nA + 255) / 256, 256, 0, stream>>>(A, Abf, nA);
  cvt_bf16_kernel<<<(8192 + 255) / 256, 256, 0, stream>>>(W0, Wb, 8192);
  cvt_bf16_kernel<<<(8192 + 255) / 256, 256, 0, stream>>>(W1, Wb + 8192, 8192);
  cvt_x_kernel<<<dim3((NB * NC) / 32, N3 / 32), 256, 0, stream>>>(x, H0);

  dim3 grid(N3 / 64, NB / 2);
  stsgcm_layer_kernel<<<grid, 128, 0, stream>>>(H0, Abf, Wb,        b0, H1, out, 0);
  stsgcm_layer_kernel<<<grid, 128, 0, stream>>>(H1, Abf, Wb + 8192, b1, H1, out, 1);
}
